// SelfAttention_44753559224634
// MI455X (gfx1250) — matrix operand order repaired and checked by reading, not compiled, not run on hardware
//
#include <hip/hip_runtime.h>
#include <hip/hip_bf16.h>

typedef __attribute__((ext_vector_type(16))) _Float16 v16h;
typedef __attribute__((ext_vector_type(8)))  _Float16 v8h;
typedef __attribute__((ext_vector_type(4)))  _Float16 v4h;
typedef __attribute__((ext_vector_type(8)))  float    v8f;

union HV { v16h v; struct { v8h lo, hi; } p; };

#define WMMA_F16(A, B, C) \
    __builtin_amdgcn_wmma_f32_16x16x32_f16(false, (A), false, (B), (short)0, (C), false, false)

static constexpr int Bn = 32, Nn = 1024, Cc = 512, Dqk = 64;
static constexpr int NCOL = 640;               // 64 q + 64 k + 512 v columns
static constexpr float SCALE = 0.04419417382415922f;  // 1/sqrt(512)

// ---------------------------------------------------------------------------
// Kernel 0: convert weights to f16, column-major Wt[col][k]; gather biases.
// ---------------------------------------------------------------------------
__global__ __launch_bounds__(256) void prep_kernel(
    const float* __restrict__ Wq, const float* __restrict__ bq,
    const float* __restrict__ Wk, const float* __restrict__ bk,
    const float* __restrict__ Wv, const float* __restrict__ bv,
    _Float16* __restrict__ Wt, float* __restrict__ biasw)
{
    int id = blockIdx.x * 256 + threadIdx.x;
    if (id >= NCOL * Cc) return;
    int col = id >> 9;          // 0..639
    int k   = id & 511;         // 0..511
    float w = (col < 64)  ? Wq[k * Dqk + col]
            : (col < 128) ? Wk[k * Dqk + (col - 64)]
                          : Wv[k * Cc + (col - 128)];
    Wt[(size_t)col * Cc + k] = (_Float16)w;
    if (k == 0)
        biasw[col] = (col < 64) ? bq[col] : (col < 128) ? bk[col - 64] : bv[col - 128];
}

// ---------------------------------------------------------------------------
// Kernel 1: QKV projection via WMMA.
//   grid (N/16, B), block 256 (8 waves). Each block: one 16-row tile of flat,
//   all 640 output columns (wave w -> column tiles w, w+8, ..., w+32).
//   Q -> Qh[b][n][64] row-major (via per-wave LDS bounce)
//   K -> Kh[b][n][64] row-major (via per-wave LDS bounce)
//   V -> Vt[b][c][n]  transposed (direct b128 stores from D layout)
// ---------------------------------------------------------------------------
__global__ __launch_bounds__(256) void qkv_kernel(
    const float* __restrict__ inp, const _Float16* __restrict__ Wt,
    const float* __restrict__ biasw,
    _Float16* __restrict__ Qh, _Float16* __restrict__ Kh, _Float16* __restrict__ Vt)
{
    __shared__ _Float16 Aup[16 * Cc];        // 16 KB: 16 rows x 512, f16
    __shared__ _Float16 bounce[8][16 * 16];  // 4 KB: per-wave transpose tile

    const int mtile = blockIdx.x, b = blockIdx.y;
    const int tid = threadIdx.x, warp = tid >> 5, lane = tid & 31;
    const int ln = lane & 15, hb = lane >> 4;   // half-wave select
    const int m0 = mtile * 16;

    // Stage 16 rows x 512 of inputs into LDS as f16 (rows are contiguous).
    const float4* src = (const float4*)(inp + ((size_t)b * Nn + m0) * Cc);
    for (int i = tid; i < 16 * Cc / 4; i += 256) {
        float4 f = src[i];
        v4h h = {(_Float16)f.x, (_Float16)f.y, (_Float16)f.z, (_Float16)f.w};
        ((v4h*)Aup)[i] = h;
    }
    __syncthreads();

    for (int t = 0; t < 5; ++t) {
        const int ct = warp + 8 * t;       // column tile 0..39
        const int c0 = ct * 16;            // concat column base
        v8f acc = {};
        const _Float16* arow = Aup + ln * Cc;
        const _Float16* wcol = Wt + (size_t)(c0 + ln) * Cc;
#pragma unroll
        for (int ks = 0; ks < 16; ++ks) {
            const int d0 = ks * 32;
            HV a;
            a.p.lo = *(const v8h*)(arow + d0 + hb * 16);      // K 0-7 / 16-23
            a.p.hi = *(const v8h*)(arow + d0 + 8 + hb * 16);  // K 8-15 / 24-31
            v16h wb = *(const v16h*)(wcol + d0 + hb * 16);    // B: 16 contig K
            acc = WMMA_F16(a.v, wb, acc);
        }
        const float bias = biasw[c0 + ln];
#pragma unroll
        for (int r = 0; r < 8; ++r) acc[r] += bias;

        if (ct < 8) {
            // Q/K tile: write D layout into bounce (lane = column)
#pragma unroll
            for (int r = 0; r < 8; ++r)
                bounce[warp][(r + hb * 8) * 16 + ln] = (_Float16)acc[r];
        } else {
            // V tile: D layout is already column-per-lane -> store transposed
            const int vc = c0 - 128 + ln;
            v8h o;
#pragma unroll
            for (int r = 0; r < 8; ++r) o[r] = (_Float16)acc[r];
            *(v8h*)(Vt + ((size_t)b * Cc + vc) * Nn + m0 + hb * 8) = o;
        }
        __syncthreads();
        if (ct < 8) {
            // Re-read row-major: lane covers row ln, half hb
            v8h o = *(const v8h*)(&bounce[warp][ln * 16 + hb * 8]);
            _Float16* dst = (ct < 4)
                ? (Qh + ((size_t)b * Nn + m0 + ln) * Dqk + ct * 16 + hb * 8)
                : (Kh + ((size_t)b * Nn + m0 + ln) * Dqk + (ct - 4) * 16 + hb * 8);
            *(v8h*)dst = o;
        }
        __syncthreads();
    }
}

// ---------------------------------------------------------------------------
// Kernel 2: flash attention per (batch, 16-row Q tile).
//   grid (N/16, B), block 256 (8 waves). 8 iterations of 128 keys.
//   Per iteration: ALL 8 waves compute one 16x16 score tile each (16x128
//   block) -> LDS; softmax parallelized over 128 threads (8 per row with
//   LDS partial max/sum reduction); then every wave rescales its
//   accumulators and does P(16x128) x V(128x64) with 16 WMMAs.
// ---------------------------------------------------------------------------
__global__ __launch_bounds__(256) void attn_kernel(
    const float* __restrict__ inp,
    const _Float16* __restrict__ Qh, const _Float16* __restrict__ Kh,
    const _Float16* __restrict__ Vt, float* __restrict__ out)
{
    __shared__ float    Ssh[16 * 128];   // 8 KB raw scores
    __shared__ _Float16 Psh[16 * 128];   // 4 KB probabilities
    __shared__ float    PMsh[128];       // per-(row,segment) partial max
    __shared__ float    PSsh[128];       // per-(row,segment) partial sum
    __shared__ float    Msh[16], Lsh[16], Csh[16];

    const int mtile = blockIdx.x, b = blockIdx.y;
    const int tid = threadIdx.x, warp = tid >> 5, lane = tid & 31;
    const int ln = lane & 15, hb = lane >> 4;
    const int m0 = mtile * 16;

    const _Float16* Qb = Qh + (size_t)b * Nn * Dqk;
    const _Float16* Kb = Kh + (size_t)b * Nn * Dqk;
    const _Float16* Vb = Vt + (size_t)b * Cc * Nn;

    if (tid < 16) { Msh[tid] = -3.0e38f; Lsh[tid] = 0.0f; }

    v8f acc[4] = {v8f{}, v8f{}, v8f{}, v8f{}};

    // Persistent Q A-fragments (d = 0..31 and 32..63); needed by every wave.
    HV qa0, qa1;
    {
        const _Float16* qrow = Qb + (size_t)(m0 + ln) * Dqk;
        qa0.p.lo = *(const v8h*)(qrow + 0  + hb * 16);
        qa0.p.hi = *(const v8h*)(qrow + 8  + hb * 16);
        qa1.p.lo = *(const v8h*)(qrow + 32 + hb * 16);
        qa1.p.hi = *(const v8h*)(qrow + 40 + hb * 16);
    }
    __syncthreads();

    for (int j = 0; j < 8; ++j) {
        // ---- scores: wave w -> keys j*128 + w*16 .. +15 ----
        {
            const int key0 = j * 128 + warp * 16;
            const _Float16* krow = Kb + (size_t)(key0 + ln) * Dqk;
            v16h kb0 = *(const v16h*)(krow + hb * 16);        // d 0..31
            v16h kb1 = *(const v16h*)(krow + 32 + hb * 16);   // d 32..63
            v8f s = {};
            s = WMMA_F16(qa0.v, kb0, s);
            s = WMMA_F16(qa1.v, kb1, s);
#pragma unroll
            for (int r = 0; r < 8; ++r)
                Ssh[(r + hb * 8) * 128 + warp * 16 + ln] = s[r] * SCALE;
        }
        __syncthreads();

        // ---- softmax phase 1: partial max (8 threads per row, 16 each) ----
        if (tid < 128) {
            const int r = tid >> 3, sg = tid & 7;
            const float* srow = Ssh + r * 128 + sg * 16;
            float pm = -3.0e38f;
#pragma unroll
            for (int k = 0; k < 16; ++k) pm = fmaxf(pm, srow[k]);
            PMsh[tid] = pm;
        }
        __syncthreads();

        // ---- softmax phase 2: exp + partial sums ----
        if (tid < 128) {
            const int r = tid >> 3, sg = tid & 7;
            float mnew = Msh[r];
#pragma unroll
            for (int s2 = 0; s2 < 8; ++s2) mnew = fmaxf(mnew, PMsh[r * 8 + s2]);
            const float* srow = Ssh + r * 128 + sg * 16;
            _Float16*   prow = Psh + r * 128 + sg * 16;
            float ps = 0.0f;
#pragma unroll
            for (int k = 0; k < 16; ++k) {
                float pexp = __expf(srow[k] - mnew);
                prow[k] = (_Float16)pexp;
                ps += pexp;
            }
            PSsh[tid] = ps;
        }
        __syncthreads();

        // ---- softmax phase 3: fold running (m, l, c) state ----
        if (tid < 16) {
            const int r = tid;
            float mold = Msh[r], mnew = mold;
#pragma unroll
            for (int s2 = 0; s2 < 8; ++s2) mnew = fmaxf(mnew, PMsh[r * 8 + s2]);
            float ls = 0.0f;
#pragma unroll
            for (int s2 = 0; s2 < 8; ++s2) ls += PSsh[r * 8 + s2];
            float csc = __expf(mold - mnew);
            Lsh[r] = Lsh[r] * csc + ls;
            Msh[r] = mnew;
            Csh[r] = csc;
        }
        __syncthreads();

        // ---- rescale accumulators, P x V ----
        float cf[8];
#pragma unroll
        for (int r = 0; r < 8; ++r) cf[r] = Csh[r + hb * 8];

        HV pa[4];
        {
            const _Float16* prow = Psh + ln * 128;
#pragma unroll
            for (int q = 0; q < 4; ++q) {
                pa[q].p.lo = *(const v8h*)(prow + q * 32 + hb * 16);
                pa[q].p.hi = *(const v8h*)(prow + q * 32 + 8 + hb * 16);
            }
        }
#pragma unroll
        for (int t = 0; t < 4; ++t) {
            const int c = warp * 64 + t * 16 + ln;
            const _Float16* vcol = Vb + (size_t)c * Nn + j * 128;
            v8f a = acc[t];
#pragma unroll
            for (int r = 0; r < 8; ++r) a[r] *= cf[r];
#pragma unroll
            for (int q = 0; q < 4; ++q) {
                v16h vb = *(const v16h*)(vcol + q * 32 + hb * 16);
                a = WMMA_F16(pa[q].v, vb, a);
            }
            acc[t] = a;
        }
        __syncthreads();
    }

    float li[8];
#pragma unroll
    for (int r = 0; r < 8; ++r) li[r] = 1.0f / Lsh[r + hb * 8];

#pragma unroll
    for (int t = 0; t < 4; ++t) {
        const int c = warp * 64 + t * 16 + ln;
#pragma unroll
        for (int r = 0; r < 8; ++r) {
            size_t idx = ((size_t)b * Nn + m0 + r + hb * 8) * Cc + c;
            out[idx] = inp[idx] + acc[t][r] * li[r];
        }
    }
}

// ---------------------------------------------------------------------------
extern "C" void kernel_launch(void* const* d_in, const int* in_sizes, int n_in,
                              void* d_out, int out_size, void* d_ws, size_t ws_size,
                              hipStream_t stream)
{
    (void)in_sizes; (void)n_in; (void)out_size; (void)ws_size;
    const float* inp = (const float*)d_in[0];
    const float* Wq  = (const float*)d_in[1];
    const float* bq  = (const float*)d_in[2];
    const float* Wk  = (const float*)d_in[3];
    const float* bk  = (const float*)d_in[4];
    const float* Wv  = (const float*)d_in[5];
    const float* bv  = (const float*)d_in[6];
    float* out = (float*)d_out;

    char* ws = (char*)d_ws;
    _Float16* Wt    = (_Float16*)ws;                               // 640*512 f16
    float*    biasw = (float*)(ws + (size_t)NCOL * Cc * 2);        // 640 f32
    char*     p     = ws + (size_t)NCOL * Cc * 2 + NCOL * 4;       // 16B aligned
    _Float16* Qh = (_Float16*)p;                 p += (size_t)Bn * Nn * Dqk * 2;
    _Float16* Kh = (_Float16*)p;                 p += (size_t)Bn * Nn * Dqk * 2;
    _Float16* Vt = (_Float16*)p;                 // Bn*Cc*Nn f16

    prep_kernel<<<(NCOL * Cc + 255) / 256, 256, 0, stream>>>(Wq, bq, Wk, bk, Wv, bv, Wt, biasw);
    qkv_kernel<<<dim3(Nn / 16, Bn), 256, 0, stream>>>(inp, Wt, biasw, Qh, Kh, Vt);
    attn_kernel<<<dim3(Nn / 16, Bn), 256, 0, stream>>>(inp, Qh, Kh, Vt, out);
}
